// VectorQuantizer_4157528343202
// MI455X (gfx1250) — compile-verified, matrix operand order checked by reading, not run on hardware
//
#include <hip/hip_runtime.h>

// ---------------------------------------------------------------------------
// VectorQuantizer for MI455X (gfx1250, wave32, WMMA).
//
// Distances d[n,k] = ||e_n||^2 + ||c_k||^2 - 2 e_n.c_k ; argmin only needs
// s[n,k] = ||c_k||^2 - 2 e_n.c_k.  The -2*e.c GEMM runs on
// v_wmma_f32_16x16x32_bf16 with split-bf16 error compensation
// (x = hi + lo; x.y ~ hi.hi + lo.hi + hi.lo), fp32 accumulation.
// ---------------------------------------------------------------------------

#define NROWS   262144
#define DDIM    256
#define KCODES  512
#define CHUNKS  8      // 256 / 32 (K per wmma)
#define NTILES  32     // 512 / 16 (codes per column tile)

typedef __bf16 bf16;
typedef __bf16 v16bf __attribute__((ext_vector_type(16)));
typedef __bf16 v8bf  __attribute__((ext_vector_type(8)));
typedef float  v8f   __attribute__((ext_vector_type(8)));

// ---------------------------------------------------------------------------
// Kernel 1: codebook -> bf16 hi/lo, per-code squared norm, zero loss accum.
// Grid: KCODES blocks x DDIM threads (block k handles code k).
// ---------------------------------------------------------------------------
__global__ __launch_bounds__(DDIM)
void vq_prep_kernel(const float* __restrict__ cb,
                    bf16* __restrict__ cb_hi, bf16* __restrict__ cb_lo,
                    float* __restrict__ cnorm, float* __restrict__ loss_accum) {
    const int k = blockIdx.x;
    const int d = threadIdx.x;
    const int i = k * DDIM + d;
    float v  = cb[i];
    bf16 h   = (bf16)v;
    cb_hi[i] = h;
    cb_lo[i] = (bf16)(v - (float)h);

    __shared__ float red[DDIM];
    red[d] = v * v;
    __syncthreads();
    #pragma unroll
    for (int s = DDIM / 2; s > 0; s >>= 1) {
        if (d < s) red[d] += red[d + s];
        __syncthreads();
    }
    if (d == 0) cnorm[k] = red[0];
    if (k == 0 && d == 0) *loss_accum = 0.0f;   // fresh every launch (graph-safe)
}

// ---------------------------------------------------------------------------
// Kernel 2: WMMA argmin.  Block = 256 threads = 8 waves; wave w owns 16 rows.
// Codebook tiles (16 codes x 256 k, bf16 hi+lo = 16KB) staged in LDS and
// shared by all 8 waves.
// ---------------------------------------------------------------------------
__global__ __launch_bounds__(256)
void vq_argmin_kernel(const int* __restrict__ x, const float* __restrict__ e,
                      const bf16* __restrict__ cb_hi, const bf16* __restrict__ cb_lo,
                      const float* __restrict__ cnorm, int* __restrict__ idx_out) {
    __shared__ bf16 lds_hi[16 * DDIM];
    __shared__ bf16 lds_lo[16 * DDIM];

    const int tid  = threadIdx.x;
    const int wave = tid >> 5;
    const int lane = tid & 31;
    const int hh   = lane >> 4;     // half-wave select (ISA A/B frag layout)
    const int l16  = lane & 15;
    const int rowbase = (blockIdx.x * 8 + wave) * 16;

    // ---- A fragments: 16 rows x 256 K of e, split bf16 hi/lo, resident. ----
    // ISA 16-bit A 16x32 layout: lanes 0-15 hold M=lane, K={c*32+0..7, +16..23};
    // lanes 16-31 hold K={c*32+8..15, +24..31}.  Element i of the v16bf maps to
    // k = c*32 + hh*8 + (i<8 ? i : i+8).
    v16bf a_hi[CHUNKS], a_lo[CHUNKS];
    {
        const float* erow = e + (size_t)(rowbase + l16) * DDIM;
        #pragma unroll
        for (int c = 0; c < CHUNKS; ++c) {
            const int kb = c * 32 + hh * 8;
            const float4* r0 = (const float4*)(erow + kb);        // k = kb..kb+7
            const float4* r1 = (const float4*)(erow + kb + 16);   // k = kb+16..kb+23
            float vals[16];
            float4 p;
            p = r0[0]; vals[0]=p.x;  vals[1]=p.y;  vals[2]=p.z;  vals[3]=p.w;
            p = r0[1]; vals[4]=p.x;  vals[5]=p.y;  vals[6]=p.z;  vals[7]=p.w;
            p = r1[0]; vals[8]=p.x;  vals[9]=p.y;  vals[10]=p.z; vals[11]=p.w;
            p = r1[1]; vals[12]=p.x; vals[13]=p.y; vals[14]=p.z; vals[15]=p.w;
            #pragma unroll
            for (int i = 0; i < 16; ++i) {
                bf16 h = (bf16)vals[i];
                a_hi[c][i] = h;
                a_lo[c][i] = (bf16)(vals[i] - (float)h);
            }
        }
    }

    // ---- Per-slot atom-type index ranges (C/D layout: slot v -> row v+8*hh,
    //      col = l16).  Ranges are inclusive [lo,hi] per the reference slices. --
    int   lo_s[8], hi_s[8], mini[8];
    float minv[8];
    #pragma unroll
    for (int v = 0; v < 8; ++v) {
        const int row = rowbase + v + 8 * hh;
        const int t   = x[2 * row];                 // x[:,0] = atom type
        int lo, hi;
        if      (t == 5) { lo = 0;   hi = 376; }    // d[:, 0:377]
        else if (t == 6) { lo = 378; hi = 432; }    // d[:, 378:433]
        else if (t == 7) { lo = 434; hi = 487; }    // d[:, 434:488]
        else             { lo = 489; hi = 510; }    // d[:, 489:511]
        lo_s[v] = lo; hi_s[v] = hi;
        minv[v] = 3.4028235e38f;
        mini[v] = lo;
    }

    // ---- Column-tile loop: stage codebook tile in LDS, 24 WMMA per tile. ----
    for (int nt = 0; nt < NTILES; ++nt) {
        __syncthreads();                            // LDS safe to overwrite
        {   // 16 codes x 256 k x 2B = 8192B per array -> 512 uint4 each
            const uint4* sh = (const uint4*)(cb_hi + nt * 16 * DDIM);
            const uint4* sl = (const uint4*)(cb_lo + nt * 16 * DDIM);
            uint4* dh = (uint4*)lds_hi;
            uint4* dl = (uint4*)lds_lo;
            #pragma unroll
            for (int j = 0; j < 2; ++j) {
                dh[tid + j * 256] = sh[tid + j * 256];
                dl[tid + j * 256] = sl[tid + j * 256];
            }
        }
        __syncthreads();

        const int   n  = nt * 16 + l16;
        const float cn = cnorm[n];

        v8f acc = {0.f, 0.f, 0.f, 0.f, 0.f, 0.f, 0.f, 0.f};
        #pragma unroll
        for (int c = 0; c < CHUNKS; ++c) {
            const int kb = c * 32 + hh * 8;
            const bf16* bh = lds_hi + l16 * DDIM + kb;
            const bf16* bl = lds_lo + l16 * DDIM + kb;
            v8bf bh0 = *(const v8bf*)(bh);
            v8bf bh1 = *(const v8bf*)(bh + 16);
            v8bf bl0 = *(const v8bf*)(bl);
            v8bf bl1 = *(const v8bf*)(bl + 16);
            v16bf b_hi = __builtin_shufflevector(bh0, bh1,
                0,1,2,3,4,5,6,7,8,9,10,11,12,13,14,15);
            v16bf b_lo = __builtin_shufflevector(bl0, bl1,
                0,1,2,3,4,5,6,7,8,9,10,11,12,13,14,15);
            // e.c ~ hi*hi + lo*hi + hi*lo (fp32 accumulate)
            acc = __builtin_amdgcn_wmma_f32_16x16x32_bf16(
                      false, a_hi[c], false, b_hi, (short)0, acc, false, false);
            acc = __builtin_amdgcn_wmma_f32_16x16x32_bf16(
                      false, a_lo[c], false, b_hi, (short)0, acc, false, false);
            acc = __builtin_amdgcn_wmma_f32_16x16x32_bf16(
                      false, a_hi[c], false, b_lo, (short)0, acc, false, false);
        }

        #pragma unroll
        for (int v = 0; v < 8; ++v) {
            const float score = cn - 2.0f * acc[v];     // ||c||^2 - 2 e.c
            const bool  inr   = (n >= lo_s[v]) && (n <= hi_s[v]);
            if (inr && score < minv[v]) { minv[v] = score; mini[v] = n; }
        }
    }

    // ---- Butterfly reduce (min, first-idx tiebreak) over the 16-lane half. --
    #pragma unroll
    for (int off = 1; off < 16; off <<= 1) {
        #pragma unroll
        for (int v = 0; v < 8; ++v) {
            float ov = __shfl_xor(minv[v], off, 32);
            int   oi = __shfl_xor(mini[v], off, 32);
            if (ov < minv[v] || (ov == minv[v] && oi < mini[v])) {
                minv[v] = ov; mini[v] = oi;
            }
        }
    }
    if (l16 == 0) {
        #pragma unroll
        for (int v = 0; v < 8; ++v)
            idx_out[rowbase + v + 8 * hh] = mini[v];
    }
}

// ---------------------------------------------------------------------------
// Kernel 3: gather + straight-through output + loss partial sums (float4).
// out = e + (q - e), matching the reference's fp arithmetic order.
// ---------------------------------------------------------------------------
__global__ __launch_bounds__(256)
void vq_gather_loss_kernel(const float* __restrict__ e, const float* __restrict__ cb,
                           const int* __restrict__ idx, float* __restrict__ out,
                           float* __restrict__ loss_accum) {
    const long long ND4 = (long long)NROWS * DDIM / 4;
    const long long stride = (long long)gridDim.x * blockDim.x;
    float lsum = 0.0f;
    for (long long i = (long long)blockIdx.x * blockDim.x + threadIdx.x;
         i < ND4; i += stride) {
        const int row = (int)(i >> 6);          // 64 float4 per row
        const int d4  = (int)(i & 63);
        const float4 q  = ((const float4*)cb)[(size_t)idx[row] * 64 + d4];
        const float4 ev = ((const float4*)e)[i];
        float4 t, o;
        t.x = q.x - ev.x; t.y = q.y - ev.y; t.z = q.z - ev.z; t.w = q.w - ev.w;
        o.x = ev.x + t.x; o.y = ev.y + t.y; o.z = ev.z + t.z; o.w = ev.w + t.w;
        ((float4*)out)[i] = o;
        lsum += t.x * t.x + t.y * t.y + t.z * t.z + t.w * t.w;
    }
    __shared__ float red[256];
    red[threadIdx.x] = lsum;
    __syncthreads();
    #pragma unroll
    for (int s = 128; s > 0; s >>= 1) {
        if ((int)threadIdx.x < s) red[threadIdx.x] += red[threadIdx.x + s];
        __syncthreads();
    }
    if (threadIdx.x == 0) atomicAdd(loss_accum, red[0]);
}

// ---------------------------------------------------------------------------
// Kernel 4: loss = (1 + 0.25) * sum / (N*D)
// ---------------------------------------------------------------------------
__global__ void vq_finalize_kernel(const float* __restrict__ loss_accum,
                                   float* __restrict__ out) {
    out[(size_t)NROWS * DDIM] = 1.25f * (*loss_accum) / (float)((long long)NROWS * DDIM);
}

// ---------------------------------------------------------------------------
extern "C" void kernel_launch(void* const* d_in, const int* in_sizes, int n_in,
                              void* d_out, int out_size, void* d_ws, size_t ws_size,
                              hipStream_t stream) {
    const int*   x  = (const int*)d_in[0];      // (N,2) ints; atom type = x[2*row]
    const float* e  = (const float*)d_in[1];    // (N,256) fp32
    const float* cb = (const float*)d_in[2];    // (512,256) fp32
    float* out = (float*)d_out;                 // N*256 quantized_st + 1 loss

    char* ws = (char*)d_ws;
    bf16*  cb_hi = (bf16*)(ws);                                   // 256 KB
    bf16*  cb_lo = (bf16*)(ws + (size_t)KCODES * DDIM * 2);       // 256 KB
    float* cnorm = (float*)(ws + (size_t)KCODES * DDIM * 4);      // 2 KB
    float* lacc  = (float*)(ws + (size_t)KCODES * DDIM * 4 + KCODES * 4);
    int*   idx   = (int*)(ws + (size_t)KCODES * DDIM * 4 + KCODES * 4 + 256); // 1 MB

    vq_prep_kernel<<<KCODES, DDIM, 0, stream>>>(cb, cb_hi, cb_lo, cnorm, lacc);
    vq_argmin_kernel<<<NROWS / 128, 256, 0, stream>>>(x, e, cb_hi, cb_lo, cnorm, idx);
    vq_gather_loss_kernel<<<4096, 256, 0, stream>>>(e, cb, idx, out, lacc);
    vq_finalize_kernel<<<1, 1, 0, stream>>>(lacc, out);
}
